// PaliGemmaWithExpertV2_79766132621535
// MI455X (gfx1250) — matrix operand order repaired and checked by reading, not compiled, not run on hardware
//
#include <hip/hip_runtime.h>

// ---------------- problem constants (from reference) ----------------
constexpr int B_   = 8;
constexpr int SP   = 512;
constexpr int SS   = 64;
constexpr int T_   = 576;          // SP + SS
constexpr int HP   = 2048;
constexpr int HS   = 1024;
constexpr int NH   = 8;            // power of two (shift 3)
constexpr int HD   = 256;          // power of two (shift 8)
constexpr int IP   = 16384;
constexpr int IS   = 4096;
constexpr int CONDN = 1024;
constexpr float EPS_   = 1e-6f;
constexpr float SCALE_ = 0.0625f;  // HD^-0.5 = 1/16

#define DEV __device__ __forceinline__

// ---------------- small helpers ----------------
DEV unsigned short f2bf(float f) {
  union { float f; unsigned int u; } c; c.f = f;
  unsigned int u = c.u + 0x7FFFu + ((c.u >> 16) & 1u);   // round-to-nearest-even
  return (unsigned short)(u >> 16);
}

DEV float gelu_tanh(float x) {
  const float k0 = 0.7978845608028654f, k1 = 0.044715f;
  float x3 = x * x * x;
  return 0.5f * x * (1.0f + tanhf(k0 * (x + k1 * x3)));
}

// ---------------- WMMA GEMM ----------------
// Computes Y = X(MxK) @ op(W) with op(W)=W^T (W is NxK, TRANSB) or W (KxN, !TRANSB).
// All M multiples of 64, N multiples of 64, K multiples of 32 in this model.
typedef __attribute__((ext_vector_type(16))) __bf16 v16bf;
typedef __attribute__((ext_vector_type(8)))  float  v8f;

union Frag {
  v16bf v;
  uint4 q[2];
  unsigned short s[16];
};

// z-dependent base offset: (z >> shift)*sA + (z & mask)*sB  (all divisors are pow2)
struct ZOff { long sA; long sB; int shift; };
DEV long zoff(ZOff o, int z) {
  return (long)(z >> o.shift) * o.sA + (long)(z & ((1 << o.shift) - 1)) * o.sB;
}

enum { EPI_NONE = 0, EPI_QKV, EPI_SCALEMASK, EPI_RES, EPI_RESGATE, EPI_GATEUP };

struct GemmP {
  const float *X, *W, *W2, *R, *G;
  float *Y;
  int M, N, K;
  int ldx, ldw, ldy, ldr;
  ZOff zx, zw, zy, zr, zg;
  float scale;
  int t0, hdShift, strideHead;   // EPI_QKV scatter params (head dim = 1<<hdShift)
};

template <int EPI, bool TRANSB, bool DUAL>
__global__ __launch_bounds__(256) void gemm_wmma_k(GemmP p) {
  constexpr int LDA  = 40;   // ushort stride, A / transposed-B tile rows (pad vs banks)
  constexpr int LDBN = 72;   // ushort stride, non-transposed B tile rows
  __shared__ __align__(16) unsigned short As[64 * LDA];
  __shared__ __align__(16) unsigned short Bs[TRANSB ? 64 * LDA : 32 * LDBN];
  __shared__ __align__(16) unsigned short Bs2[DUAL ? 64 * LDA : 16];

  const int z = blockIdx.z;
  const float* X  = p.X + zoff(p.zx, z);
  const float* W  = p.W + zoff(p.zw, z);
  const float* W2 = DUAL ? (p.W2 + zoff(p.zw, z)) : nullptr;
  float* Y        = p.Y + zoff(p.zy, z);
  const float* R  = (EPI == EPI_SCALEMASK || EPI == EPI_RES || EPI == EPI_RESGATE)
                    ? (p.R + zoff(p.zr, z)) : nullptr;
  const float* G  = (EPI == EPI_RESGATE) ? (p.G + zoff(p.zg, z)) : nullptr;

  const int tid  = threadIdx.x;
  const int lane = tid & 31;
  const int wave = tid >> 5;
  const int wm   = wave & 3;    // 4 M-subtiles of 16 -> 64
  const int wn0  = wave >> 2;   // 0..1; wave handles nsub wn0 and wn0+2

  const int m0 = blockIdx.x * 64;
  const int n0 = blockIdx.y * 64;

  v8f acc0 = {0.f, 0.f, 0.f, 0.f, 0.f, 0.f, 0.f, 0.f};
  v8f acc1 = acc0, acc2 = acc0, acc3 = acc0;

  for (int k0 = 0; k0 < p.K; k0 += 32) {
    __syncthreads();
    {   // stage A tile (64 x 32) as bf16; float4 loads -> global_load_b128
      const int r = tid >> 2, c = (tid & 3) * 8;
      const float* src = X + (long)(m0 + r) * p.ldx + (k0 + c);
      float4 f0 = *reinterpret_cast<const float4*>(src);
      float4 f1 = *reinterpret_cast<const float4*>(src + 4);
      if (k0 + 32 < p.K) __builtin_prefetch(src + 32, 0, 3);
      unsigned short* dst = &As[r * LDA + c];
      dst[0] = f2bf(f0.x); dst[1] = f2bf(f0.y); dst[2] = f2bf(f0.z); dst[3] = f2bf(f0.w);
      dst[4] = f2bf(f1.x); dst[5] = f2bf(f1.y); dst[6] = f2bf(f1.z); dst[7] = f2bf(f1.w);
    }
    if (TRANSB) {   // stage W tile (rows = N, 64 x 32)
      const int r = tid >> 2, c = (tid & 3) * 8;
      const float* src = W + (long)(n0 + r) * p.ldw + (k0 + c);
      float4 f0 = *reinterpret_cast<const float4*>(src);
      float4 f1 = *reinterpret_cast<const float4*>(src + 4);
      if (k0 + 32 < p.K) __builtin_prefetch(src + 32, 0, 3);
      unsigned short* dst = &Bs[r * LDA + c];
      dst[0] = f2bf(f0.x); dst[1] = f2bf(f0.y); dst[2] = f2bf(f0.z); dst[3] = f2bf(f0.w);
      dst[4] = f2bf(f1.x); dst[5] = f2bf(f1.y); dst[6] = f2bf(f1.z); dst[7] = f2bf(f1.w);
      if (DUAL) {
        const float* src2 = W2 + (long)(n0 + r) * p.ldw + (k0 + c);
        float4 g0 = *reinterpret_cast<const float4*>(src2);
        float4 g1 = *reinterpret_cast<const float4*>(src2 + 4);
        if (k0 + 32 < p.K) __builtin_prefetch(src2 + 32, 0, 3);
        unsigned short* dst2 = &Bs2[r * LDA + c];
        dst2[0] = f2bf(g0.x); dst2[1] = f2bf(g0.y); dst2[2] = f2bf(g0.z); dst2[3] = f2bf(g0.w);
        dst2[4] = f2bf(g1.x); dst2[5] = f2bf(g1.y); dst2[6] = f2bf(g1.z); dst2[7] = f2bf(g1.w);
      }
    } else {        // stage W tile (rows = K, 32 x 64)
      const int r = tid >> 3, c = (tid & 7) * 8;
      const float* src = W + (long)(k0 + r) * p.ldw + (n0 + c);
      float4 f0 = *reinterpret_cast<const float4*>(src);
      float4 f1 = *reinterpret_cast<const float4*>(src + 4);
      if (k0 + 32 < p.K) __builtin_prefetch(src + (long)32 * p.ldw, 0, 3);
      unsigned short* dst = &Bs[r * LDBN + c];
      dst[0] = f2bf(f0.x); dst[1] = f2bf(f0.y); dst[2] = f2bf(f0.z); dst[3] = f2bf(f0.w);
      dst[4] = f2bf(f1.x); dst[5] = f2bf(f1.y); dst[6] = f2bf(f1.z); dst[7] = f2bf(f1.w);
    }
    __syncthreads();

    // A fragment: lane L: M = L&15, K = (L>>4)*16 + elem(0..15)
    Frag a;
    {
      const int row = wm * 16 + (lane & 15);
      const int kb  = (lane >> 4) * 16;
      a.q[0] = *reinterpret_cast<const uint4*>(&As[row * LDA + kb]);
      a.q[1] = *reinterpret_cast<const uint4*>(&As[row * LDA + kb + 8]);
    }

#pragma unroll
    for (int s = 0; s < 2; ++s) {
      const int nsub = wn0 + 2 * s;
      // B fragment: lane L: N = L&15, K = (L>>4)*16 + elem(0..15)
      Frag b;
      if (TRANSB) {
        const int row = nsub * 16 + (lane & 15);
        const int kb  = (lane >> 4) * 16;
        b.q[0] = *reinterpret_cast<const uint4*>(&Bs[row * LDA + kb]);
        b.q[1] = *reinterpret_cast<const uint4*>(&Bs[row * LDA + kb + 8]);
      } else {
        const int col = nsub * 16 + (lane & 15);
        const int kb  = (lane >> 4) * 16;
#pragma unroll
        for (int i = 0; i < 16; ++i) b.s[i] = Bs[(kb + i) * LDBN + col];
      }
      v8f& acc = s ? acc1 : acc0;
#if defined(__HIP_DEVICE_COMPILE__)
      acc = __builtin_amdgcn_wmma_f32_16x16x32_bf16(false, a.v, false, b.v,
                                                    (short)0, acc, false, false);
#endif
      if (DUAL) {
        Frag b2;
        const int row = nsub * 16 + (lane & 15);
        const int kb  = (lane >> 4) * 16;
        b2.q[0] = *reinterpret_cast<const uint4*>(&Bs2[row * LDA + kb]);
        b2.q[1] = *reinterpret_cast<const uint4*>(&Bs2[row * LDA + kb + 8]);
        v8f& accu = s ? acc3 : acc2;
#if defined(__HIP_DEVICE_COMPILE__)
        accu = __builtin_amdgcn_wmma_f32_16x16x32_bf16(false, a.v, false, b2.v,
                                                       (short)0, accu, false, false);
#endif
      }
    }
  }

  // C/D fragment: element r: M = r + (lane>>4)*8, N = lane&15
#pragma unroll
  for (int s = 0; s < 2; ++s) {
    const int nsub = wn0 + 2 * s;
    const v8f& acc  = s ? acc1 : acc0;
    const v8f& accu = s ? acc3 : acc2;
#pragma unroll
    for (int r = 0; r < 8; ++r) {
      const int m = m0 + wm * 16 + (lane >> 4) * 8 + r;
      const int n = n0 + nsub * 16 + (lane & 15);
      float v = acc[r];
      if (EPI == EPI_GATEUP) v = gelu_tanh(v) * accu[r];
      if (EPI == EPI_QKV) {
        const int hdm = (1 << p.hdShift) - 1;
        Y[(long)(n >> p.hdShift) * p.strideHead
          + ((long)(p.t0 + m) << p.hdShift) + (n & hdm)] = v;
      } else if (EPI == EPI_SCALEMASK) {
        Y[(long)m * p.ldy + n] = v * p.scale + R[(long)m * p.ldr + n];
      } else if (EPI == EPI_RES) {
        Y[(long)m * p.ldy + n] = v + R[(long)m * p.ldr + n];
      } else if (EPI == EPI_RESGATE) {
        Y[(long)m * p.ldy + n] = R[(long)m * p.ldr + n] + v * G[n];
      } else {
        Y[(long)m * p.ldy + n] = v;
      }
    }
  }
}

// ---------------- elementwise / reduction kernels ----------------
__global__ __launch_bounds__(256) void rms_scale_k(const float* __restrict__ x,
                                                   const float* __restrict__ w,
                                                   float* __restrict__ y, int len) {
  const float* xr = x + (long)blockIdx.x * len;
  float* yr = y + (long)blockIdx.x * len;
  __shared__ float red[8];
  const int tid = threadIdx.x;
  float ss = 0.f;
  for (int i = tid; i < len; i += 256) { float v = xr[i]; ss += v * v; }
#pragma unroll
  for (int o = 16; o > 0; o >>= 1) ss += __shfl_xor(ss, o, 32);
  if ((tid & 31) == 0) red[tid >> 5] = ss;
  __syncthreads();
  float tot = 0.f;
#pragma unroll
  for (int i = 0; i < 8; ++i) tot += red[i];
  float inv = rsqrtf(tot / (float)len + EPS_);
  for (int i = tid; i < len; i += 256) yr[i] = xr[i] * inv * (1.0f + w[i]);
}

// tokens_per_b is a power of two -> pass its shift
__global__ __launch_bounds__(256) void ada_norm_k(const float* __restrict__ x,
                                                  const float* __restrict__ mod,
                                                  float* __restrict__ y,
                                                  int len, int tokShift) {
  const int row = blockIdx.x;
  const int b = row >> tokShift;
  const float* xr = x + (long)row * len;
  float* yr = y + (long)row * len;
  const float* sc = mod + (long)b * 3 * len;
  const float* sh = sc + len;
  __shared__ float red[8];
  const int tid = threadIdx.x;
  float ss = 0.f;
  for (int i = tid; i < len; i += 256) { float v = xr[i]; ss += v * v; }
#pragma unroll
  for (int o = 16; o > 0; o >>= 1) ss += __shfl_xor(ss, o, 32);
  if ((tid & 31) == 0) red[tid >> 5] = ss;
  __syncthreads();
  float tot = 0.f;
#pragma unroll
  for (int i = 0; i < 8; ++i) tot += red[i];
  float inv = rsqrtf(tot / (float)len + EPS_);
  for (int i = tid; i < len; i += 256)
    yr[i] = xr[i] * inv * (1.0f + sc[i]) + sh[i];
}

// mod = cond @ W^T + b   (tiny: 8 x 3072 x 1024)
__global__ __launch_bounds__(256) void ada_mod_k(const float* __restrict__ cond,
                                                 const float* __restrict__ w,
                                                 const float* __restrict__ bias,
                                                 float* __restrict__ mod, int N, int K) {
  const int n = blockIdx.x * 256 + threadIdx.x;
  const int b = blockIdx.y;
  if (n >= N) return;
  const float* c  = cond + (long)b * K;
  const float* wr = w + (long)n * K;
  float s = bias[n];
  for (int k = 0; k < K; k += 4) {
    float4 cv = *reinterpret_cast<const float4*>(c + k);
    float4 wv = *reinterpret_cast<const float4*>(wr + k);
    s += cv.x * wv.x + cv.y * wv.y + cv.z * wv.z + cv.w * wv.w;
  }
  mod[(long)b * N + n] = s;
}

// grid: (T_*HD/2 / 256, B_ << headShift); x layout (B, heads, T, HD)
__global__ __launch_bounds__(256) void rope_k(float* __restrict__ x,
                                              const int* __restrict__ pos,
                                              int headShift) {
  const int idx = blockIdx.x * 256 + threadIdx.x;   // 0 .. T_*HD/2-1
  const int j = idx & (HD / 2 - 1);
  const int t = idx >> 7;                           // HD/2 = 128
  const long bh = blockIdx.y;
  const int b = (int)(bh >> headShift);
  float p = (float)pos[(long)b * T_ + t];
  float inv_freq = __expf(-__logf(10000.0f) * (float)(2 * j) / (float)HD);
  float ang = p * inv_freq;
  float s, c;
  __sincosf(ang, &s, &c);
  float* base = x + (bh * T_ + t) * (long)HD;
  float x1 = base[j], x2 = base[j + HD / 2];
  base[j]          = x1 * c - x2 * s;
  base[j + HD / 2] = x2 * c + x1 * s;
}

__global__ __launch_bounds__(256) void softmax_k(float* __restrict__ data, int len) {
  float* row = data + (long)blockIdx.x * len;
  __shared__ float red[8];
  const int tid = threadIdx.x;
  float mx = -3.4e38f;
  for (int i = tid; i < len; i += 256) mx = fmaxf(mx, row[i]);
#pragma unroll
  for (int o = 16; o > 0; o >>= 1) mx = fmaxf(mx, __shfl_xor(mx, o, 32));
  if ((tid & 31) == 0) red[tid >> 5] = mx;
  __syncthreads();
  float m = red[0];
#pragma unroll
  for (int i = 1; i < 8; ++i) m = fmaxf(m, red[i]);
  __syncthreads();
  float sum = 0.f;
  for (int i = tid; i < len; i += 256) { float e = __expf(row[i] - m); row[i] = e; sum += e; }
#pragma unroll
  for (int o = 16; o > 0; o >>= 1) sum += __shfl_xor(sum, o, 32);
  if ((tid & 31) == 0) red[tid >> 5] = sum;
  __syncthreads();
  float s = 0.f;
#pragma unroll
  for (int i = 0; i < 8; ++i) s += red[i];
  float inv = 1.0f / s;
  for (int i = tid; i < len; i += 256) row[i] *= inv;
}

// ---------------- host orchestration ----------------
static inline GemmP mkP() {
  GemmP p;
  p.X = p.W = p.W2 = p.R = p.G = nullptr; p.Y = nullptr;
  p.M = p.N = p.K = 0;
  p.ldx = p.ldw = p.ldy = p.ldr = 1;
  p.zx = {0, 0, 0}; p.zw = {0, 0, 0}; p.zy = {0, 0, 0};
  p.zr = {0, 0, 0}; p.zg = {0, 0, 0};
  p.scale = 1.f; p.t0 = 0; p.hdShift = 0; p.strideHead = 0;
  return p;
}

constexpr int NH_SHIFT = 3;   // NH = 8
constexpr int HD_SHIFT = 8;   // HD = 256
constexpr int SS_SHIFT = 6;   // SS = 64

extern "C" void kernel_launch(void* const* d_in, const int* in_sizes, int n_in,
                              void* d_out, int out_size, void* d_ws, size_t ws_size,
                              hipStream_t stream) {
  (void)in_sizes; (void)n_in; (void)out_size; (void)ws_size;
  const float* x_prefix = (const float*)d_in[0];
  const float* x_suffix = (const float*)d_in[1];
  const float* cond     = (const float*)d_in[2];
  const float* amask    = (const float*)d_in[3];
  const float* p_ln1_w  = (const float*)d_in[4];
  const float* p_q      = (const float*)d_in[5];
  const float* p_k      = (const float*)d_in[6];
  const float* p_v      = (const float*)d_in[7];
  const float* p_o      = (const float*)d_in[8];
  const float* p_ln2_w  = (const float*)d_in[9];
  const float* p_gate   = (const float*)d_in[10];
  const float* p_up     = (const float*)d_in[11];
  const float* p_down   = (const float*)d_in[12];
  const float* s_ada1_w = (const float*)d_in[13];
  const float* s_ada1_b = (const float*)d_in[14];
  const float* s_q      = (const float*)d_in[15];
  const float* s_k      = (const float*)d_in[16];
  const float* s_v      = (const float*)d_in[17];
  const float* s_o      = (const float*)d_in[18];
  const float* s_ada2_w = (const float*)d_in[19];
  const float* s_ada2_b = (const float*)d_in[20];
  const float* s_gate   = (const float*)d_in[21];
  const float* s_up     = (const float*)d_in[22];
  const float* s_down   = (const float*)d_in[23];
  const int*   pos_ids  = (const int*)d_in[24];

  float* out = (float*)d_out;
  float* out_prefix = out;
  float* out_suffix = out + (long)B_ * SP * HP;

  // workspace partition
  float* ws = (float*)d_ws;
  float* qb     = ws; ws += (long)B_ * NH * T_ * HD;   // (B,NH,T,HD)
  float* k0b    = ws; ws += (long)B_ * T_ * HD;        // (B,T,HD)
  float* v0b    = ws; ws += (long)B_ * T_ * HD;
  float* hp     = ws; ws += (long)B_ * SP * HP;        // normed prefix (reused as h)
  float* hsb    = ws; ws += (long)B_ * SS * HS;        // normed suffix (reused as h2)
  float* mod1   = ws; ws += (long)B_ * 3 * HS;
  float* mod2   = ws; ws += (long)B_ * 3 * HS;
  float* scores = ws; ws += (long)B_ * NH * T_ * T_;
  float* attnb  = ws; ws += (long)B_ * T_ * NH * HD;   // (B,T,NH*HD)
  float* resp   = ws; ws += (long)B_ * SP * HP;
  float* ress   = ws; ws += (long)B_ * SS * HS;
  float* interp = ws; ws += (long)B_ * SP * IP;
  float* inters = ws; ws += (long)B_ * SS * IS;

  // 1) hp = rms(x_prefix) * (1 + p_ln1_w)
  rms_scale_k<<<dim3(B_ * SP), 256, 0, stream>>>(x_prefix, p_ln1_w, hp, HP);
  // 2) mod1 = cond @ s_ada1_w^T + s_ada1_b ; 3) hs = ada_norm(x_suffix, mod1)
  ada_mod_k<<<dim3((3 * HS + 255) / 256, B_), 256, 0, stream>>>(cond, s_ada1_w, s_ada1_b, mod1, 3 * HS, CONDN);
  ada_norm_k<<<dim3(B_ * SS), 256, 0, stream>>>(x_suffix, mod1, hsb, HS, SS_SHIFT);

  // 4) prefix QKV (scatter into (B,NH,T,HD) / (B,T,HD))
  {
    GemmP p = mkP();
    p.X = hp; p.ldx = HP; p.zx = {(long)SP * HP, 0, 0};
    p.ldw = HP; p.M = SP; p.K = HP;
    p.hdShift = HD_SHIFT; p.strideHead = T_ * HD; p.t0 = 0;
    p.W = p_q; p.N = NH * HD; p.Y = qb; p.zy = {(long)NH * T_ * HD, 0, 0};
    gemm_wmma_k<EPI_QKV, true, false><<<dim3(SP / 64, (NH * HD) / 64, B_), 256, 0, stream>>>(p);
    p.N = HD;
    p.W = p_k; p.Y = k0b; p.zy = {(long)T_ * HD, 0, 0};
    gemm_wmma_k<EPI_QKV, true, false><<<dim3(SP / 64, HD / 64, B_), 256, 0, stream>>>(p);
    p.W = p_v; p.Y = v0b;
    gemm_wmma_k<EPI_QKV, true, false><<<dim3(SP / 64, HD / 64, B_), 256, 0, stream>>>(p);
  }
  // 5) suffix QKV (t0 = SP)
  {
    GemmP p = mkP();
    p.X = hsb; p.ldx = HS; p.zx = {(long)SS * HS, 0, 0};
    p.ldw = HS; p.M = SS; p.K = HS;
    p.hdShift = HD_SHIFT; p.strideHead = T_ * HD; p.t0 = SP;
    p.W = s_q; p.N = NH * HD; p.Y = qb; p.zy = {(long)NH * T_ * HD, 0, 0};
    gemm_wmma_k<EPI_QKV, true, false><<<dim3(SS / 64, (NH * HD) / 64, B_), 256, 0, stream>>>(p);
    p.N = HD;
    p.W = s_k; p.Y = k0b; p.zy = {(long)T_ * HD, 0, 0};
    gemm_wmma_k<EPI_QKV, true, false><<<dim3(SS / 64, HD / 64, B_), 256, 0, stream>>>(p);
    p.W = s_v; p.Y = v0b;
    gemm_wmma_k<EPI_QKV, true, false><<<dim3(SS / 64, HD / 64, B_), 256, 0, stream>>>(p);
  }
  // 6) RoPE on q and k0
  rope_k<<<dim3((T_ * HD / 2) / 256, B_ * NH), 256, 0, stream>>>(qb, pos_ids, NH_SHIFT);
  rope_k<<<dim3((T_ * HD / 2) / 256, B_), 256, 0, stream>>>(k0b, pos_ids, 0);
  // 7) scores = q @ k0^T * SCALE + mask     (z = b*NH+h)
  {
    GemmP p = mkP();
    p.X = qb;  p.ldx = HD; p.zx = {0, (long)T_ * HD, 31};   // z*T*HD (shift 31 -> z>>31==0)
    p.X = qb;  p.zx = {(long)T_ * HD, 0, 0};
    p.W = k0b; p.ldw = HD; p.zw = {(long)T_ * HD, 0, NH_SHIFT};
    p.Y = scores; p.ldy = T_; p.zy = {(long)T_ * T_, 0, 0};
    p.R = amask;  p.ldr = T_; p.zr = {(long)T_ * T_, 0, NH_SHIFT};
    p.M = T_; p.N = T_; p.K = HD; p.scale = SCALE_;
    gemm_wmma_k<EPI_SCALEMASK, true, false><<<dim3(T_ / 64, T_ / 64, B_ * NH), 256, 0, stream>>>(p);
  }
  // 8) softmax rows
  softmax_k<<<dim3(B_ * NH * T_), 256, 0, stream>>>(scores, T_);
  // 9) attn = probs @ v0 -> scatter to (B,T,NH*HD)
  {
    GemmP p = mkP();
    p.X = scores; p.ldx = T_; p.zx = {(long)T_ * T_, 0, 0};
    p.W = v0b; p.ldw = HD; p.zw = {(long)T_ * HD, 0, NH_SHIFT};
    p.Y = attnb; p.ldy = NH * HD; p.zy = {(long)T_ * NH * HD, (long)HD, NH_SHIFT};
    p.M = T_; p.N = HD; p.K = T_;
    gemm_wmma_k<EPI_NONE, false, false><<<dim3(T_ / 64, HD / 64, B_ * NH), 256, 0, stream>>>(p);
  }
  // 10) res_p = x_prefix + a_p @ p_o^T
  {
    GemmP p = mkP();
    p.X = attnb; p.ldx = NH * HD; p.zx = {(long)T_ * NH * HD, 0, 0};
    p.W = p_o; p.ldw = NH * HD;
    p.R = x_prefix; p.ldr = HP; p.zr = {(long)SP * HP, 0, 0};
    p.Y = resp; p.ldy = HP; p.zy = {(long)SP * HP, 0, 0};
    p.M = SP; p.N = HP; p.K = NH * HD;
    gemm_wmma_k<EPI_RES, true, false><<<dim3(SP / 64, HP / 64, B_), 256, 0, stream>>>(p);
  }
  // 11) res_s = x_suffix + (a_s @ s_o^T) * gate1
  {
    GemmP p = mkP();
    p.X = attnb + (long)SP * NH * HD; p.ldx = NH * HD; p.zx = {(long)T_ * NH * HD, 0, 0};
    p.W = s_o; p.ldw = NH * HD;
    p.R = x_suffix; p.ldr = HS; p.zr = {(long)SS * HS, 0, 0};
    p.G = mod1 + 2 * HS; p.zg = {(long)3 * HS, 0, 0};
    p.Y = ress; p.ldy = HS; p.zy = {(long)SS * HS, 0, 0};
    p.M = SS; p.N = HS; p.K = NH * HD;
    gemm_wmma_k<EPI_RESGATE, true, false><<<dim3(SS / 64, HS / 64, B_), 256, 0, stream>>>(p);
  }
  // 12) h = rms(res_p) * (1 + p_ln2_w)   (reuse hp)
  rms_scale_k<<<dim3(B_ * SP), 256, 0, stream>>>(resp, p_ln2_w, hp, HP);
  // 13) inter_p = gelu(h @ p_gate^T) * (h @ p_up^T)
  {
    GemmP p = mkP();
    p.X = hp; p.ldx = HP;
    p.W = p_gate; p.W2 = p_up; p.ldw = HP;
    p.Y = interp; p.ldy = IP;
    p.M = B_ * SP; p.N = IP; p.K = HP;
    gemm_wmma_k<EPI_GATEUP, true, true><<<dim3((B_ * SP) / 64, IP / 64, 1), 256, 0, stream>>>(p);
  }
  // 14) out_prefix = res_p + inter_p @ p_down^T
  {
    GemmP p = mkP();
    p.X = interp; p.ldx = IP;
    p.W = p_down; p.ldw = IP;
    p.R = resp; p.ldr = HP;
    p.Y = out_prefix; p.ldy = HP;
    p.M = B_ * SP; p.N = HP; p.K = IP;
    gemm_wmma_k<EPI_RES, true, false><<<dim3((B_ * SP) / 64, HP / 64, 1), 256, 0, stream>>>(p);
  }
  // 15-16) mod2 / h2 = ada_norm(res_s, mod2)   (reuse hsb)
  ada_mod_k<<<dim3((3 * HS + 255) / 256, B_), 256, 0, stream>>>(cond, s_ada2_w, s_ada2_b, mod2, 3 * HS, CONDN);
  ada_norm_k<<<dim3(B_ * SS), 256, 0, stream>>>(ress, mod2, hsb, HS, SS_SHIFT);
  // 17) inter_s = gelu(h2 @ s_gate^T) * (h2 @ s_up^T)
  {
    GemmP p = mkP();
    p.X = hsb; p.ldx = HS;
    p.W = s_gate; p.W2 = s_up; p.ldw = HS;
    p.Y = inters; p.ldy = IS;
    p.M = B_ * SS; p.N = IS; p.K = HS;
    gemm_wmma_k<EPI_GATEUP, true, true><<<dim3((B_ * SS) / 64, IS / 64, 1), 256, 0, stream>>>(p);
  }
  // 18) out_suffix = res_s + (inter_s @ s_down^T) * gate2
  {
    GemmP p = mkP();
    p.X = inters; p.ldx = IS; p.zx = {(long)SS * IS, 0, 0};
    p.W = s_down; p.ldw = IS;
    p.R = ress; p.ldr = HS; p.zr = {(long)SS * HS, 0, 0};
    p.G = mod2 + 2 * HS; p.zg = {(long)3 * HS, 0, 0};
    p.Y = out_suffix; p.ldy = HS; p.zy = {(long)SS * HS, 0, 0};
    p.M = SS; p.N = HS; p.K = IS;
    gemm_wmma_k<EPI_RESGATE, true, false><<<dim3(SS / 64, HS / 64, B_), 256, 0, stream>>>(p);
  }
}
